// MultiHeadAttention_53429393162906
// MI455X (gfx1250) — matrix operand order repaired and checked by reading, not compiled, not run on hardware
//
#include <hip/hip_runtime.h>
#include <stdint.h>

#define B_  2
#define S_  2048
#define DM_ 2048
#define H_  16
#define DH_ 128
#define M_  (B_ * S_)   // 4096 rows total

// Padded LDS row: 40 bf16 = 80 bytes. Multiple of 16B (b128 fragment loads
// stay aligned), and 20-dword stride -> 16 row-parallel lanes map to 16
// distinct banks (gcd(20,64)=4, 16 consecutive multiples of 20 mod 64 distinct).
#define LDSP 40

typedef __attribute__((ext_vector_type(16))) __bf16 v16bf;
typedef __attribute__((ext_vector_type(8)))  __bf16 v8bf;
typedef __attribute__((ext_vector_type(8)))  float  v8f;

// ---------------------------------------------------------------------------
// WMMA helpers (CDNA5 bf16, 16x16x32, f32 accumulate)
// ---------------------------------------------------------------------------
__device__ __forceinline__ v8f wmma_bf16(v16bf a, v16bf b, v8f c) {
  return __builtin_amdgcn_wmma_f32_16x16x32_bf16(false, a, false, b, (short)0, c,
                                                 false, false);
}

// A-matrix fragment (16x32 row-major): lane holds row (lane&15); element runs
// K = [half*8 .. +7] and K = [16+half*8 .. +7]  -> two 16-byte loads.
__device__ __forceinline__ v16bf load_afrag(const __bf16* rowbase, int lh) {
  v16bf a;
  ((v8bf*)&a)[0] = *(const v8bf*)(rowbase + lh * 8);
  ((v8bf*)&a)[1] = *(const v8bf*)(rowbase + 16 + lh * 8);
  return a;
}

// B-matrix fragment (32x16, K-contiguous): lane holds col (lane&15);
// elements K = [half*8 .. +7] and K = [16+half*8 .. +7] -> two 16-byte loads.
__device__ __forceinline__ v16bf load_bfrag(const __bf16* rowbase, int lh) {
  v16bf b;
  ((v8bf*)&b)[0] = *(const v8bf*)(rowbase + lh * 8);
  ((v8bf*)&b)[1] = *(const v8bf*)(rowbase + 16 + lh * 8);
  return b;
}

// ---------------------------------------------------------------------------
// Kernel 1: QKV projection, double-buffered.
// X[4096,2048] f32 x W[h][2048,128] f32 + bias -> bf16,
// TRANS=0: [B,H,S,DH]; TRANS=1: [B,H,DH,S].
// 256 thr (8 waves), 128x128 block tile, 32x64 wave tile (2x4 WMMA tiles).
// ---------------------------------------------------------------------------
template <int TRANS>
__global__ __launch_bounds__(256) void proj_gemm_kernel(
    const float* __restrict__ X, const float* __restrict__ W,
    const float* __restrict__ bias, __bf16* __restrict__ Y) {
  __shared__ __attribute__((aligned(16))) __bf16 As[2][128][LDSP];
  __shared__ __attribute__((aligned(16))) __bf16 Bs[2][128][LDSP];

  const int tid  = threadIdx.x;
  const int wave = tid >> 5, lane = tid & 31;
  const int lh = lane >> 4, ln = lane & 15;
  const int bm = blockIdx.x * 128, bn = blockIdx.y * 128;
  const int wm = (wave >> 1) * 32, wn = (wave & 1) * 64;
  const int h  = bn >> 7;                       // one head per column tile
  const float* Wh = W + (size_t)h * DM_ * DH_;  // [2048][128] f32

  // Per-thread staging coordinates (4 float4 each for A and B per K-step).
  const int cA  = (4 * tid) & 31;   // A col; rows rA0 + 32*j
  const int rA0 = (4 * tid) >> 5;
  const int nB  = (4 * tid) & 127;  // B n;  k rows kB0 + 8*j
  const int kB0 = (4 * tid) >> 7;

  float4 ra[4], rb[4];
  auto loadNext = [&](int k0) {
#pragma unroll
    for (int j = 0; j < 4; ++j) {
      ra[j] = *(const float4*)&X[(size_t)(bm + rA0 + 32 * j) * DM_ + k0 + cA];
      rb[j] = *(const float4*)&Wh[(size_t)(k0 + kB0 + 8 * j) * DH_ + nB];
    }
  };
  auto storeStage = [&](int buf) {
#pragma unroll
    for (int j = 0; j < 4; ++j) {
      const int r = rA0 + 32 * j;
      As[buf][r][cA + 0] = (__bf16)ra[j].x; As[buf][r][cA + 1] = (__bf16)ra[j].y;
      As[buf][r][cA + 2] = (__bf16)ra[j].z; As[buf][r][cA + 3] = (__bf16)ra[j].w;
      const int kk = kB0 + 8 * j;
      Bs[buf][nB + 0][kk] = (__bf16)rb[j].x; Bs[buf][nB + 1][kk] = (__bf16)rb[j].y;
      Bs[buf][nB + 2][kk] = (__bf16)rb[j].z; Bs[buf][nB + 3][kk] = (__bf16)rb[j].w;
    }
  };

  v8f acc[2][4] = {};

  loadNext(0);
  storeStage(0);
  __syncthreads();

  int cur = 0;
  for (int k0 = 0; k0 < DM_; k0 += 32) {
    const bool pf = (k0 + 32 < DM_);
    if (pf) loadNext(k0 + 32);  // global loads in flight during WMMAs

    v16bf af[2], bfr[4];
#pragma unroll
    for (int i = 0; i < 2; ++i) af[i]  = load_afrag(&As[cur][wm + i * 16 + ln][0], lh);
#pragma unroll
    for (int j = 0; j < 4; ++j) bfr[j] = load_bfrag(&Bs[cur][wn + j * 16 + ln][0], lh);
#pragma unroll
    for (int i = 0; i < 2; ++i)
#pragma unroll
      for (int j = 0; j < 4; ++j) acc[i][j] = wmma_bf16(af[i], bfr[j], acc[i][j]);

    if (pf) storeStage(cur ^ 1);
    __syncthreads();
    cur ^= 1;
  }

  // Epilogue: bias + bf16 store.  C layout: row = v + 8*half, col = ln.
#pragma unroll
  for (int i = 0; i < 2; ++i)
#pragma unroll
    for (int j = 0; j < 4; ++j) {
      const int e = wn + j * 16 + ln;  // 0..127 within head
      const float bv = bias[(size_t)h * DH_ + e];
#pragma unroll
      for (int v = 0; v < 8; ++v) {
        const int m  = bm + wm + i * 16 + 8 * lh + v;
        const int bb = m >> 11, s = m & (S_ - 1);
        const float val = acc[i][j][v] + bv;
        if constexpr (!TRANS)
          Y[(((size_t)bb * H_ + h) * S_ + s) * DH_ + e] = (__bf16)val;
        else
          Y[(((size_t)bb * H_ + h) * DH_ + e) * S_ + s] = (__bf16)val;
      }
    }
}

// ---------------------------------------------------------------------------
// Kernel 2: flash attention.  Each wave owns 16 Q rows; streams K/V in
// 32-column chunks with online softmax.  Q/K in [B,H,S,DH] bf16; V transposed
// [B,H,DH,S] bf16.  Output written in concat layout [B,S,H*DH] bf16.
// ---------------------------------------------------------------------------
__global__ __launch_bounds__(256) void attn_kernel(
    const __bf16* __restrict__ qs, const __bf16* __restrict__ ks,
    const __bf16* __restrict__ vt, __bf16* __restrict__ outc) {
  __shared__ __attribute__((aligned(16))) __bf16 Pl[8][16][LDSP];

  const int tid  = threadIdx.x;
  const int wave = tid >> 5, lane = tid & 31;
  const int lh = lane >> 4, ln = lane & 15;
  const int bh = blockIdx.y;
  const int b  = bh >> 4, h = bh & (H_ - 1);
  const int q0 = blockIdx.x * 128 + wave * 16;

  const __bf16* Q  = qs + ((size_t)bh * S_ + q0) * DH_;
  const __bf16* K  = ks + (size_t)bh * S_ * DH_;
  const __bf16* Vt = vt + (size_t)bh * DH_ * S_;

  v16bf qf[4];
#pragma unroll
  for (int t = 0; t < 4; ++t)
    qf[t] = load_afrag(Q + (size_t)ln * DH_ + t * 32, lh);

  v8f o[4] = {};
  v8f mrun, lrun;
#pragma unroll
  for (int v = 0; v < 8; ++v) { mrun[v] = -1.0e30f; lrun[v] = 0.0f; }

  const float scale = 0.088388347648318447f;  // 1/sqrt(128)

  for (int c0 = 0; c0 < S_; c0 += 32) {
    v8f sc[2] = {};
#pragma unroll
    for (int u = 0; u < 2; ++u)
#pragma unroll
      for (int t = 0; t < 4; ++t) {
        const v16bf kf =
            load_bfrag(K + (size_t)(c0 + u * 16 + ln) * DH_ + t * 32, lh);
        sc[u] = wmma_bf16(qf[t], kf, sc[u]);
      }

    v8f pmax;
#pragma unroll
    for (int v = 0; v < 8; ++v) {
      sc[0][v] *= scale; sc[1][v] *= scale;
      pmax[v] = fmaxf(sc[0][v], sc[1][v]);
    }
#pragma unroll
    for (int off = 1; off < 16; off <<= 1)
#pragma unroll
      for (int v = 0; v < 8; ++v)
        pmax[v] = fmaxf(pmax[v], __shfl_xor(pmax[v], off, 32));

    v8f alpha, psum;
#pragma unroll
    for (int v = 0; v < 8; ++v) {
      const float mnew = fmaxf(mrun[v], pmax[v]);
      alpha[v] = __expf(mrun[v] - mnew);
      mrun[v]  = mnew;
      const float p0 = __expf(sc[0][v] - mnew);
      const float p1 = __expf(sc[1][v] - mnew);
      sc[0][v] = p0; sc[1][v] = p1;
      psum[v]  = p0 + p1;
    }
#pragma unroll
    for (int off = 1; off < 16; off <<= 1)
#pragma unroll
      for (int v = 0; v < 8; ++v) psum[v] += __shfl_xor(psum[v], off, 32);
#pragma unroll
    for (int v = 0; v < 8; ++v) lrun[v] = lrun[v] * alpha[v] + psum[v];
#pragma unroll
    for (int t = 0; t < 4; ++t)
#pragma unroll
      for (int v = 0; v < 8; ++v) o[t][v] *= alpha[v];

    // P: C layout -> row-major 16x32 bf16 in LDS, reload as A fragment.
    // Per-wave LDS ops complete in order; no barrier needed.
#pragma unroll
    for (int u = 0; u < 2; ++u)
#pragma unroll
      for (int v = 0; v < 8; ++v)
        Pl[wave][8 * lh + v][u * 16 + ln] = (__bf16)sc[u][v];

    const v16bf pf = load_afrag(&Pl[wave][ln][0], lh);

#pragma unroll
    for (int t = 0; t < 4; ++t) {
      const v16bf vf = load_bfrag(Vt + (size_t)(t * 16 + ln) * S_ + c0, lh);
      o[t] = wmma_bf16(pf, vf, o[t]);
    }
  }

#pragma unroll
  for (int t = 0; t < 4; ++t)
#pragma unroll
    for (int v = 0; v < 8; ++v) {
      const int srow = q0 + 8 * lh + v;
      const float val = o[t][v] / lrun[v];
      outc[((size_t)b * S_ + srow) * DM_ + h * DH_ + t * 16 + ln] = (__bf16)val;
    }
}

// ---------------------------------------------------------------------------
// Kernel 3: output projection, double-buffered.
// concat[4096,2048] bf16 x W0[2048,2048] f32 + b0 -> f32 d_out.
// A tile is a raw bf16 copy: staged with GLOBAL_LOAD_ASYNC_TO_LDS_B128
// (ASYNCcnt-tracked DMA; LDS dest = low 32 bits of the shared address),
// fenced with s_wait_asynccnt before the tile barrier.
// ---------------------------------------------------------------------------
__global__ __launch_bounds__(256) void out_gemm_kernel(
    const __bf16* __restrict__ A, const float* __restrict__ W0,
    const float* __restrict__ b0, float* __restrict__ Y) {
  __shared__ __attribute__((aligned(16))) __bf16 As[2][128][LDSP];
  __shared__ __attribute__((aligned(16))) __bf16 Bs[2][128][LDSP];

  const int tid  = threadIdx.x;
  const int wave = tid >> 5, lane = tid & 31;
  const int lh = lane >> 4, ln = lane & 15;
  const int bm = blockIdx.x * 128, bn = blockIdx.y * 128;
  const int wm = (wave >> 1) * 32, wn = (wave & 1) * 64;

  // A staging: 2 x 16B async copies per thread (e = 8*tid + 2048*j).
  const int cA  = (8 * tid) & 31;   // 0,8,16,24
  const int rA0 = (8 * tid) >> 5;   // rows rA0 + 64*j
  // B staging: 4 float4 per thread.
  const int nB  = (4 * tid) & 127;
  const int kB0 = (4 * tid) >> 7;

  auto asyncA = [&](int buf, int k0) {
#pragma unroll
    for (int j = 0; j < 2; ++j) {
      const int r = rA0 + 64 * j;
      const unsigned long long ga =
          (unsigned long long)(uintptr_t)&A[(size_t)(bm + r) * DM_ + k0 + cA];
      const unsigned ldsoff = (unsigned)(uintptr_t)&As[buf][r][cA];
      asm volatile("global_load_async_to_lds_b128 %0, %1, off"
                   :: "v"(ldsoff), "v"(ga) : "memory");
    }
  };

  float4 rb[4];
  auto loadB = [&](int k0) {
#pragma unroll
    for (int j = 0; j < 4; ++j)
      rb[j] = *(const float4*)&W0[(size_t)(k0 + kB0 + 8 * j) * DM_ + bn + nB];
  };
  auto storeB = [&](int buf) {
#pragma unroll
    for (int j = 0; j < 4; ++j) {
      const int kk = kB0 + 8 * j;
      Bs[buf][nB + 0][kk] = (__bf16)rb[j].x; Bs[buf][nB + 1][kk] = (__bf16)rb[j].y;
      Bs[buf][nB + 2][kk] = (__bf16)rb[j].z; Bs[buf][nB + 3][kk] = (__bf16)rb[j].w;
    }
  };

  v8f acc[2][4] = {};

  asyncA(0, 0);
  loadB(0);
  storeB(0);
  asm volatile("s_wait_asynccnt 0x0" ::: "memory");
  __syncthreads();

  int cur = 0;
  for (int k0 = 0; k0 < DM_; k0 += 32) {
    const bool pf = (k0 + 32 < DM_);
    if (pf) {
      asyncA(cur ^ 1, k0 + 32);  // DMA next A tile during WMMAs
      loadB(k0 + 32);
    }

    v16bf af[2], bfr[4];
#pragma unroll
    for (int i = 0; i < 2; ++i) af[i]  = load_afrag(&As[cur][wm + i * 16 + ln][0], lh);
#pragma unroll
    for (int j = 0; j < 4; ++j) bfr[j] = load_bfrag(&Bs[cur][wn + j * 16 + ln][0], lh);
#pragma unroll
    for (int i = 0; i < 2; ++i)
#pragma unroll
      for (int j = 0; j < 4; ++j) acc[i][j] = wmma_bf16(af[i], bfr[j], acc[i][j]);

    if (pf) storeB(cur ^ 1);
    asm volatile("s_wait_asynccnt 0x0" ::: "memory");
    __syncthreads();
    cur ^= 1;
  }

#pragma unroll
  for (int i = 0; i < 2; ++i)
#pragma unroll
    for (int j = 0; j < 4; ++j) {
      const int col = bn + wn + j * 16 + ln;
      const float bv = b0[col];
#pragma unroll
      for (int v = 0; v < 8; ++v) {
        const int m = bm + wm + i * 16 + 8 * lh + v;
        Y[(size_t)m * DM_ + col] = acc[i][j][v] + bv;
      }
    }
}

// ---------------------------------------------------------------------------
// Launch
// ---------------------------------------------------------------------------
extern "C" void kernel_launch(void* const* d_in, const int* in_sizes, int n_in,
                              void* d_out, int out_size, void* d_ws,
                              size_t ws_size, hipStream_t stream) {
  (void)in_sizes; (void)n_in; (void)out_size; (void)ws_size;
  const float* q  = (const float*)d_in[0];
  const float* k  = (const float*)d_in[1];
  const float* v  = (const float*)d_in[2];
  const float* Wq = (const float*)d_in[3];
  const float* bq = (const float*)d_in[4];
  const float* Wk = (const float*)d_in[5];
  const float* bk = (const float*)d_in[6];
  const float* Wv = (const float*)d_in[7];
  const float* bv = (const float*)d_in[8];
  const float* W0 = (const float*)d_in[9];
  const float* b0 = (const float*)d_in[10];
  float* out = (float*)d_out;

  const size_t nHead = (size_t)B_ * H_ * S_ * DH_;  // 8.39M elems
  __bf16* qs  = (__bf16*)d_ws;
  __bf16* ksb = qs + nHead;
  __bf16* vt  = ksb + nHead;
  __bf16* cc  = vt + nHead;   // concat heads [B,S,DM] bf16

  dim3 blk(256);
  dim3 ggrid(M_ / 128, DM_ / 128);  // 32 x 16
  proj_gemm_kernel<0><<<ggrid, blk, 0, stream>>>(q, Wq, bq, qs);
  proj_gemm_kernel<0><<<ggrid, blk, 0, stream>>>(k, Wk, bk, ksb);
  proj_gemm_kernel<1><<<ggrid, blk, 0, stream>>>(v, Wv, bv, vt);
  attn_kernel<<<dim3(S_ / 128, B_ * H_), blk, 0, stream>>>(qs, ksb, vt, cc);
  out_gemm_kernel<<<ggrid, blk, 0, stream>>>(cc, W0, b0, out);
}
